// Block_77352361001115
// MI455X (gfx1250) — matrix operand order repaired and checked by reading, not compiled, not run on hardware
//
#include <hip/hip_runtime.h>
#include <cmath>
#include <cstdint>

// ---------------------------------------------------------------------------
// AFNO block for MI455X (gfx1250): all GEMM-shaped work on
// v_wmma_f32_16x16x32_bf16 (fp32 accumulate), DFTs as dense matmuls,
// async global->LDS staging (ASYNCcnt) with double-buffered LDS pipeline,
// 128x64 block tile, 32x64 per-wave tile (2 A-frags x 4 B-frags).
// ---------------------------------------------------------------------------

typedef __bf16 bf16;
typedef __attribute__((ext_vector_type(16))) __bf16 bf16x16;
typedef __attribute__((ext_vector_type(8)))  __bf16 bf16x8;
typedef __attribute__((ext_vector_type(8)))  float  f32x8;

static constexpr int kB  = 2;
static constexpr int kH  = 90;
static constexpr int kW  = 180;
static constexpr int kC  = 768;
static constexpr int kWF = 46;            // kept rfft modes along W
static constexpr int kT  = kB * kH * kW;  // 32400 tokens
static constexpr int kHID = 3072;
static constexpr long kFreqB = (long)kH * kWF * kC;   // per-b stride in freq layout
static constexpr int  kBS = 96;           // block size (C / 8)

#define USE_ASYNC_LDS 1

#if USE_ASYNC_LDS
__device__ __forceinline__ void asyncLoadB128(unsigned ldsOff, const void* src) {
  asm volatile("global_load_async_to_lds_b128 %0, %1, off"
               :: "v"(ldsOff), "v"((unsigned long long)(uintptr_t)src)
               : "memory");
}
__device__ __forceinline__ void waitAsync() {
#if __has_builtin(__builtin_amdgcn_s_wait_asynccnt)
  __builtin_amdgcn_s_wait_asynccnt(0);
#else
  asm volatile("s_wait_asynccnt 0x0" ::: "memory");
#endif
}
#else
__device__ __forceinline__ void waitAsync() {}
#endif

// ------------------------------- GEMM ---------------------------------------
// C[M,N] = epi( alpha * A[M,K] @ op(B) (+ Cacc) (+ bias) ) (+ res)
// A row-major (lda). TB=0: B row-major [N][K]; TB=1: B stored [K][N] (ldb = K-row stride).
// Two-level batch: z -> (z / inner, z % inner) with outer/inner strides.

struct GemmP {
  const bf16* A; long lda, sAo, sAi;
  const bf16* B; long ldb, sBo, sBi;
  void* Out;     long ldc, sCo, sCi;
  const float* Cacc;        // fp32, same ldc/strides as Out
  const float* bias;        // bias[z*biasStride + n]
  long biasStride;
  const float* res;         // fp32 residual, same ldc (batch==1 only)
  float alpha;
  int outBf16, epi;         // epi: 0 none, 1 relu, 2 softshrink(0.01), 3 gelu(exact)
  int M, N, K, batchInner;
  int vecA, vecB;           // 16B-aligned fast staging allowed
};

#define TM 128
#define TN 64
#define KT 64
#define LDSK 72   // row stride (bf16): 144B = 16B-aligned, 36-bank step (no conflicts)

template <int TB>
__global__ __launch_bounds__(128) void gemm_bf16_wmma(GemmP p) {
  __shared__ __attribute__((aligned(16))) bf16 As[2][TM][LDSK];
  __shared__ __attribute__((aligned(16))) bf16 Bs[2][TN][LDSK];

  const int z  = blockIdx.z;
  const int zb = z / p.batchInner, zi = z % p.batchInner;
  const bf16* A  = p.A + (long)zb * p.sAo + (long)zi * p.sAi;
  const bf16* Bp = p.B + (long)zb * p.sBo + (long)zi * p.sBi;
  const long cbase = (long)zb * p.sCo + (long)zi * p.sCi;

  const int m0 = blockIdx.y * TM;
  const int n0 = blockIdx.x * TN;
  const int t = threadIdx.x;
  const int wave = t >> 5, lane = t & 31;
  const int lh = lane & 15, sel = lane >> 4;
  const bool interior = (m0 + TM <= p.M) && (n0 + TN <= p.N);

  f32x8 acc[2][4];
#pragma unroll
  for (int i = 0; i < 2; ++i)
#pragma unroll
    for (int j = 0; j < 4; ++j)
#pragma unroll
      for (int r = 0; r < 8; ++r) acc[i][j][r] = 0.f;

  const int ktiles = (p.K + KT - 1) / KT;

  // -------- staging of one KT-wide tile into LDS buffer `buf` --------------
  auto stage = [&](int kt, int buf) {
    const int k0 = kt * KT;
    const bool full = interior && (k0 + KT <= p.K);
    // ---- A tile: 128 rows x 64 k ----
    if (full) {
#pragma unroll
      for (int g = 0; g < 8; ++g) {
        const int gid = t + 128 * g;
        const int r = gid >> 3, cg = gid & 7;
        const bf16* s = A + (long)(m0 + r) * p.lda + k0 + cg * 8;
        if (p.vecA) {
#if USE_ASYNC_LDS
          asyncLoadB128((unsigned)(uintptr_t)&As[buf][r][cg * 8], s);
#else
          *(bf16x8*)&As[buf][r][cg * 8] = *(const bf16x8*)s;
#endif
        } else {
          bf16x8 v;
#pragma unroll
          for (int e = 0; e < 8; ++e) v[e] = s[e];
          *(bf16x8*)&As[buf][r][cg * 8] = v;
        }
      }
    } else {
#pragma unroll
      for (int g = 0; g < 8; ++g) {
        const int gid = t + 128 * g;
        const int r = gid >> 3, cg = gid & 7;
        const int gm = m0 + r;
        const int gmc = gm < p.M ? gm : p.M - 1;
        bf16x8 v;
#pragma unroll
        for (int e = 0; e < 8; ++e) {
          const int gk = k0 + cg * 8 + e;
          const int gkc = gk < p.K ? gk : p.K - 1;
          const bf16 val = A[(long)gmc * p.lda + gkc];
          v[e] = (gm < p.M && gk < p.K) ? val : (bf16)0.f;
        }
        *(bf16x8*)&As[buf][r][cg * 8] = v;
      }
    }
    // ---- B tile: 64 n-rows x 64 k ----
    if (TB == 0) {
      if (full) {
#pragma unroll
        for (int g = 0; g < 4; ++g) {
          const int gid = t + 128 * g;
          const int r = gid >> 3, cg = gid & 7;
          const bf16* s = Bp + (long)(n0 + r) * p.ldb + k0 + cg * 8;
          if (p.vecB) {
#if USE_ASYNC_LDS
            asyncLoadB128((unsigned)(uintptr_t)&Bs[buf][r][cg * 8], s);
#else
            *(bf16x8*)&Bs[buf][r][cg * 8] = *(const bf16x8*)s;
#endif
          } else {
            bf16x8 v;
#pragma unroll
            for (int e = 0; e < 8; ++e) v[e] = s[e];
            *(bf16x8*)&Bs[buf][r][cg * 8] = v;
          }
        }
      } else {
#pragma unroll
        for (int g = 0; g < 4; ++g) {
          const int gid = t + 128 * g;
          const int r = gid >> 3, cg = gid & 7;
          const int gn = n0 + r;
          const int gnc = gn < p.N ? gn : p.N - 1;
          bf16x8 v;
#pragma unroll
          for (int e = 0; e < 8; ++e) {
            const int gk = k0 + cg * 8 + e;
            const int gkc = gk < p.K ? gk : p.K - 1;
            const bf16 val = Bp[(long)gnc * p.ldb + gkc];
            v[e] = (gn < p.N && gk < p.K) ? val : (bf16)0.f;
          }
          *(bf16x8*)&Bs[buf][r][cg * 8] = v;
        }
      }
    } else {
      // B stored [K][N]: read columns (coalesced along n), write K-contiguous LDS rows
      const int n = t & 63, kh = (t >> 6) * 32;
      if (full) {
        const bf16* s = Bp + (long)(k0 + kh) * p.ldb + n0 + n;
        if (k0 + KT < p.K) __builtin_prefetch(s + (long)KT * p.ldb, 0, 1);
#pragma unroll
        for (int h = 0; h < 4; ++h) {
          bf16x8 v;
#pragma unroll
          for (int e = 0; e < 8; ++e) v[e] = s[(long)(h * 8 + e) * p.ldb];
          *(bf16x8*)&Bs[buf][n][kh + h * 8] = v;
        }
      } else {
        const int gn = n0 + n;
        const int gnc = gn < p.N ? gn : p.N - 1;
#pragma unroll
        for (int h = 0; h < 4; ++h) {
          bf16x8 v;
#pragma unroll
          for (int e = 0; e < 8; ++e) {
            const int gk = k0 + kh + h * 8 + e;
            const int gkc = gk < p.K ? gk : p.K - 1;
            const bf16 val = Bp[(long)gkc * p.ldb + gnc];
            v[e] = (gn < p.N && gk < p.K) ? val : (bf16)0.f;
          }
          *(bf16x8*)&Bs[buf][n][kh + h * 8] = v;
        }
      }
    }
  };

  // -------- compute one KT tile from LDS buffer `buf` ----------------------
  auto compute = [&](int buf) {
    const int ab = sel ? 8 : 0;
    const int bb = sel ? 8 : 0;
#pragma unroll
    for (int ks = 0; ks < KT; ks += 32) {
      bf16x16 afrag[2];
#pragma unroll
      for (int i = 0; i < 2; ++i) {
        const int arow = wave * 32 + i * 16 + lh;
        const bf16x8 alo = *(const bf16x8*)&As[buf][arow][ks + ab];
        const bf16x8 ahi = *(const bf16x8*)&As[buf][arow][ks + ab + 16];
#pragma unroll
        for (int e = 0; e < 8; ++e) { afrag[i][e] = alo[e]; afrag[i][8 + e] = ahi[e]; }
      }
#pragma unroll
      for (int j = 0; j < 4; ++j) {
        const int bn = j * 16 + lh;
        const bf16x8 blo = *(const bf16x8*)&Bs[buf][bn][ks + bb];
        const bf16x8 bhi = *(const bf16x8*)&Bs[buf][bn][ks + bb + 16];
        bf16x16 bfrag;
#pragma unroll
        for (int e = 0; e < 8; ++e) { bfrag[e] = blo[e]; bfrag[8 + e] = bhi[e]; }
        acc[0][j] = __builtin_amdgcn_wmma_f32_16x16x32_bf16(
            false, afrag[0], false, bfrag, (short)0, acc[0][j], false, false);
        acc[1][j] = __builtin_amdgcn_wmma_f32_16x16x32_bf16(
            false, afrag[1], false, bfrag, (short)0, acc[1][j], false, false);
      }
    }
  };

  // -------- pipelined main loop: stage next tile while computing current ---
  stage(0, 0);
  for (int kt = 0; kt < ktiles; ++kt) {
    const int buf = kt & 1;
    waitAsync();          // own async LDS writes done
    __syncthreads();      // everyone's tile visible
    if (kt + 1 < ktiles) stage(kt + 1, buf ^ 1);
    compute(buf);
  }

  // ---------------------------- epilogue ------------------------------------
  float* outF = (float*)p.Out;
  bf16*  outH = (bf16*)p.Out;
#pragma unroll
  for (int i = 0; i < 2; ++i) {
#pragma unroll
    for (int j = 0; j < 4; ++j) {
#pragma unroll
      for (int r = 0; r < 8; ++r) {
        const int m = m0 + wave * 32 + i * 16 + (sel ? 8 : 0) + r;
        const int n = n0 + j * 16 + lh;
        if (m < p.M && n < p.N) {
          const long idx = cbase + (long)m * p.ldc + n;
          float v = p.alpha * acc[i][j][r];
          if (p.Cacc) v += p.Cacc[idx];
          if (p.bias) v += p.bias[(long)z * p.biasStride + n];
          if (p.epi == 1)      v = v > 0.f ? v : 0.f;
          else if (p.epi == 2) { float a = fabsf(v) - 0.01f; v = a > 0.f ? copysignf(a, v) : 0.f; }
          else if (p.epi == 3) v = 0.5f * v * (1.f + erff(v * 0.70710678118f));
          if (p.res) v += p.res[idx];
          if (p.outBf16) outH[idx] = (bf16)v; else outF[idx] = v;
        }
      }
    }
  }
}

// ------------------------- elementwise kernels ------------------------------

__device__ __forceinline__ float blockReduceSum(float v) {
  __shared__ float s[8];
#pragma unroll
  for (int o = 16; o > 0; o >>= 1) v += __shfl_xor(v, o, 32);
  const int lane = threadIdx.x & 31, w = threadIdx.x >> 5;
  if (lane == 0) s[w] = v;
  __syncthreads();
  v = (threadIdx.x < 8) ? s[threadIdx.x] : 0.f;
  if (w == 0) {
#pragma unroll
    for (int o = 4; o > 0; o >>= 1) v += __shfl_xor(v, o, 32);
    if (lane == 0) s[0] = v;
  }
  __syncthreads();
  const float r = s[0];
  __syncthreads();
  return r;
}

// LayerNorm over C=768, fp32 in -> bf16 out
__global__ __launch_bounds__(256) void k_ln(const float* __restrict__ x,
                                            const float* __restrict__ w,
                                            const float* __restrict__ b,
                                            bf16* __restrict__ out) {
  const long row = blockIdx.x;
  const float* xr = x + row * kC;
  float v[3], s = 0.f, q = 0.f;
#pragma unroll
  for (int u = 0; u < 3; ++u) {
    v[u] = xr[threadIdx.x + u * 256];
    s += v[u]; q += v[u] * v[u];
  }
  s = blockReduceSum(s);
  q = blockReduceSum(q);
  const float mu = s * (1.f / kC);
  const float rs = rsqrtf(q * (1.f / kC) - mu * mu + 1e-5f);
#pragma unroll
  for (int u = 0; u < 3; ++u) {
    const int c = threadIdx.x + u * 256;
    out[row * kC + c] = (bf16)((v[u] - mu) * rs * w[c] + b[c]);
  }
}

// residual2 = irfft + LN1(x) + x ; then LN2 -> bf16
__global__ __launch_bounds__(256) void k_fuse_mid(const float* __restrict__ x,
                                                  const float* __restrict__ afno,
                                                  const bf16* __restrict__ ln1,
                                                  const float* __restrict__ w2,
                                                  const float* __restrict__ b2,
                                                  float* __restrict__ resid_out,
                                                  bf16* __restrict__ ln2out) {
  const long row = blockIdx.x;
  const long base = row * kC;
  float v[3], s = 0.f, q = 0.f;
#pragma unroll
  for (int u = 0; u < 3; ++u) {
    const int c = threadIdx.x + u * 256;
    const float tv = afno[base + c] + (float)ln1[base + c] + x[base + c];
    v[u] = tv; s += tv; q += tv * tv;
  }
  s = blockReduceSum(s);
  q = blockReduceSum(q);
  const float mu = s * (1.f / kC);
  const float rs = rsqrtf(q * (1.f / kC) - mu * mu + 1e-5f);
#pragma unroll
  for (int u = 0; u < 3; ++u) {
    const int c = threadIdx.x + u * 256;
    resid_out[base + c] = v[u];
    ln2out[base + c] = (bf16)((v[u] - mu) * rs * w2[c] + b2[c]);
  }
}

// transpose+convert all weights to bf16 [N][K] layout
__global__ void k_prep(const float* __restrict__ w1, const float* __restrict__ w2,
                       const float* __restrict__ fc1w, const float* __restrict__ fc2w,
                       bf16* w1t, bf16* w2t, bf16* fc1t, bf16* fc2t) {
  const int i = blockIdx.x * 256 + threadIdx.x;
  const int W1N = 2 * 8 * kBS * kBS;  // 147456
  const int FCN = kC * kHID;          // 2359296
  if (i < W1N) {
    int s = i / 73728, r = i % 73728, k = r / 9216, r2 = r % 9216, o = r2 / 96, ii = r2 % 96;
    w1t[i] = (bf16)w1[((long)((s * 8 + k) * 96 + ii)) * 96 + o];
  } else if (i < 2 * W1N) {
    int d = i - W1N;
    int s = d / 73728, r = d % 73728, k = r / 9216, r2 = r % 9216, o = r2 / 96, ii = r2 % 96;
    w2t[d] = (bf16)w2[((long)((s * 8 + k) * 96 + ii)) * 96 + o];
  } else if (i < 2 * W1N + FCN) {
    int d = i - 2 * W1N; int o = d / kC, ii = d % kC;
    fc1t[d] = (bf16)fc1w[(long)ii * kHID + o];
  } else if (i < 2 * W1N + 2 * FCN) {
    int d = i - 2 * W1N - FCN; int o = d / kHID, ii = d % kHID;
    fc2t[d] = (bf16)fc2w[(long)ii * kC + o];
  }
}

// DFT matrices (ortho-normalized, phases reduced mod N for fp32 accuracy)
__global__ void k_dft(bf16* dwr, bf16* dwi, bf16* er, bf16* ei,
                      bf16* gr, bf16* gi, bf16* cr, bf16* ci) {
  const int i = blockIdx.x * 256 + threadIdx.x;
  const float PI2 = 6.283185307179586f;
  const float SW = 0.07453559924999299f;  // 1/sqrt(180)
  const float SH = 0.10540925533894598f;  // 1/sqrt(90)
  if (i < kWF * kW) {                       // forward rfft along W: 46 x 180
    int wf = i / kW, w = i % kW;
    int ph = (wf * w) % kW;
    float th = PI2 * ph / (float)kW;
    dwr[i] = (bf16)(cosf(th) * SW);
    dwi[i] = (bf16)(-sinf(th) * SW);
  } else if (i < kWF * kW + kH * kH) {      // fft / ifft along H: 90 x 90
    int d = i - kWF * kW;
    int hp = d / kH, h = d % kH;
    int ph = (hp * h) % kH;
    float th = PI2 * ph / (float)kH;
    er[d] = (bf16)(cosf(th) * SH);
    ei[d] = (bf16)(-sinf(th) * SH);
    gr[d] = (bf16)(cosf(th) * SH);
    gi[d] = (bf16)(sinf(th) * SH);
  } else if (i < kWF * kW + kH * kH + kW * kWF) { // irfft along W: 180 x 46
    int d = i - kWF * kW - kH * kH;
    int w = d / kWF, wf = d % kWF;
    float a = (wf == 0) ? 1.f : 2.f;        // Hermitian fold (wf<46 < Nyquist)
    int ph = (w * wf) % kW;
    float th = PI2 * ph / (float)kW;
    cr[d] = (bf16)(a * cosf(th) * SW);
    ci[d] = (bf16)(-a * sinf(th) * SW);
  }
}

// ------------------------------ host side -----------------------------------

static inline void gemm(hipStream_t st,
                        const bf16* A, long lda, long sAo, long sAi,
                        const bf16* B, long ldb, long sBo, long sBi, int TB,
                        void* Out, int outBf16, long ldc, long sCo, long sCi,
                        const float* Cacc, const float* bias, long biasStride,
                        const float* res, float alpha, int epi,
                        int M, int N, int K, int batch, int inner) {
  GemmP p{A, lda, sAo, sAi, B, ldb, sBo, sBi, Out, ldc, sCo, sCi,
          Cacc, bias, biasStride, res, alpha, outBf16, epi, M, N, K, inner,
          /*vecA*/ (int)(lda % 8 == 0 && sAo % 8 == 0 && sAi % 8 == 0),
          /*vecB*/ (int)(!TB && ldb % 8 == 0 && sBo % 8 == 0 && sBi % 8 == 0)};
  dim3 g((N + TN - 1) / TN, (M + TM - 1) / TM, batch);
  if (TB) gemm_bf16_wmma<1><<<g, dim3(128), 0, st>>>(p);
  else    gemm_bf16_wmma<0><<<g, dim3(128), 0, st>>>(p);
}

extern "C" void kernel_launch(void* const* d_in, const int* in_sizes, int n_in,
                              void* d_out, int out_size, void* d_ws, size_t ws_size,
                              hipStream_t stream) {
  (void)in_sizes; (void)n_in; (void)out_size; (void)ws_size;
  const float* x    = (const float*)d_in[0];
  const float* n1w  = (const float*)d_in[1];
  const float* n1b  = (const float*)d_in[2];
  const float* w1   = (const float*)d_in[3];
  const float* b1   = (const float*)d_in[4];
  const float* w2   = (const float*)d_in[5];
  const float* b2   = (const float*)d_in[6];
  const float* n2w  = (const float*)d_in[7];
  const float* n2b  = (const float*)d_in[8];
  const float* fc1w = (const float*)d_in[9];
  const float* fc1b = (const float*)d_in[10];
  const float* fc2w = (const float*)d_in[11];
  const float* fc2b = (const float*)d_in[12];
  float* out = (float*)d_out;

  char* ws = (char*)d_ws;
  size_t cur = 0;
  auto alloc = [&](size_t bytes) -> char* {
    char* p = ws + cur;
    cur = (cur + bytes + 255) & ~(size_t)255;
    return p;
  };

  const long FE = (long)kB * kH * kWF * kC;  // 6,359,040 freq elements

  bf16* w1t  = (bf16*)alloc(2 * 8 * 96 * 96 * sizeof(bf16));
  bf16* w2t  = (bf16*)alloc(2 * 8 * 96 * 96 * sizeof(bf16));
  bf16* fc1t = (bf16*)alloc((size_t)kC * kHID * sizeof(bf16));
  bf16* fc2t = (bf16*)alloc((size_t)kC * kHID * sizeof(bf16));
  bf16* dwr  = (bf16*)alloc(kWF * kW * sizeof(bf16));
  bf16* dwi  = (bf16*)alloc(kWF * kW * sizeof(bf16));
  bf16* er   = (bf16*)alloc(kH * kH * sizeof(bf16));
  bf16* ei   = (bf16*)alloc(kH * kH * sizeof(bf16));
  bf16* gr   = (bf16*)alloc(kH * kH * sizeof(bf16));
  bf16* gi   = (bf16*)alloc(kH * kH * sizeof(bf16));
  bf16* cr   = (bf16*)alloc(kW * kWF * sizeof(bf16));
  bf16* ci   = (bf16*)alloc(kW * kWF * sizeof(bf16));
  bf16* ln1bf = (bf16*)alloc((size_t)kT * kC * sizeof(bf16));
  bf16* fXr  = (bf16*)alloc(FE * sizeof(bf16));  // X after rfft_w; later o1r; later Zr
  bf16* fXi  = (bf16*)alloc(FE * sizeof(bf16));
  bf16* fYr  = (bf16*)alloc(FE * sizeof(bf16));  // Y after fft_h; later o2r
  bf16* fYi  = (bf16*)alloc(FE * sizeof(bf16));
  float* tmp = (float*)alloc(FE * sizeof(float));
  float* afno = (float*)alloc((size_t)kT * kC * sizeof(float));
  bf16* ln2bf = (bf16*)alloc((size_t)kT * kC * sizeof(bf16));
  bf16* h1    = (bf16*)alloc((size_t)kT * kHID * sizeof(bf16));

  // ---- prep: weights + DFT matrices (deterministic, regenerated each call)
  k_prep<<<dim3((2 * 147456 + 2 * 2359296 + 255) / 256), dim3(256), 0, stream>>>(
      w1, w2, fc1w, fc2w, w1t, w2t, fc1t, fc2t);
  k_dft<<<dim3((kWF * kW + kH * kH + kW * kWF + 255) / 256), dim3(256), 0, stream>>>(
      dwr, dwi, er, ei, gr, gi, cr, ci);

  // ---- LN1
  k_ln<<<dim3(kT), dim3(256), 0, stream>>>(x, n1w, n1b, ln1bf);

  // strides in the freq layout [b][h][46][768]
  const long sWC = (long)kWF * kC;      // 35328
  const long sBfreq = kFreqB;           // 3,179,520
  const long sRow = (long)kW * kC;      // 138240  ([b][h] row of spatial tensor)

  // ---- rfft along W:  X = D @ ln1   per (b,h);  M=46 N=768 K=180
  gemm(stream, dwr, kW, 0, 0, ln1bf, kC, 90 * sRow, sRow, 1,
       fXr, 1, kC, sBfreq, sWC, nullptr, nullptr, 0, nullptr, 1.f, 0,
       kWF, kC, kW, kB * kH, kH);
  gemm(stream, dwi, kW, 0, 0, ln1bf, kC, 90 * sRow, sRow, 1,
       fXi, 1, kC, sBfreq, sWC, nullptr, nullptr, 0, nullptr, 1.f, 0,
       kWF, kC, kW, kB * kH, kH);

  // ---- fft along H (complex):  Y = E @ X  per (b,wf);  M=90 N=768 K=90
  gemm(stream, er, kH, 0, 0, fXr, sWC, sBfreq, kC, 1,
       tmp, 0, sWC, sBfreq, kC, nullptr, nullptr, 0, nullptr, 1.f, 0,
       kH, kC, kH, kB * kWF, kWF);
  gemm(stream, ei, kH, 0, 0, fXi, sWC, sBfreq, kC, 1,
       fYr, 1, sWC, sBfreq, kC, tmp, nullptr, 0, nullptr, -1.f, 0,
       kH, kC, kH, kB * kWF, kWF);
  gemm(stream, er, kH, 0, 0, fXi, sWC, sBfreq, kC, 1,
       tmp, 0, sWC, sBfreq, kC, nullptr, nullptr, 0, nullptr, 1.f, 0,
       kH, kC, kH, kB * kWF, kWF);
  gemm(stream, ei, kH, 0, 0, fXr, sWC, sBfreq, kC, 1,
       fYi, 1, sWC, sBfreq, kC, tmp, nullptr, 0, nullptr, 1.f, 0,
       kH, kC, kH, kB * kWF, kWF);

  // ---- Fourier MLP layer 1 (per-block 96x96, relu):  M=8280 N=96 K=96
  const int MT = kB * kH * kWF;  // 8280 freq tokens
  // o1r = relu(Yr@W1r - Yi@W1i + b1r)   -> fXr
  gemm(stream, fYr, kC, 0, kBS, w1t, kBS, 0, kBS * kBS, 0,
       tmp, 0, kC, 0, kBS, nullptr, nullptr, 0, nullptr, 1.f, 0,
       MT, kBS, kBS, 8, 8);
  gemm(stream, fYi, kC, 0, kBS, w1t + 8 * kBS * kBS, kBS, 0, kBS * kBS, 0,
       fXr, 1, kC, 0, kBS, tmp, b1, kBS, nullptr, -1.f, 1,
       MT, kBS, kBS, 8, 8);
  // o1i = relu(Yi@W1r + Yr@W1i + b1i)   -> fXi
  gemm(stream, fYi, kC, 0, kBS, w1t, kBS, 0, kBS * kBS, 0,
       tmp, 0, kC, 0, kBS, nullptr, nullptr, 0, nullptr, 1.f, 0,
       MT, kBS, kBS, 8, 8);
  gemm(stream, fYr, kC, 0, kBS, w1t + 8 * kBS * kBS, kBS, 0, kBS * kBS, 0,
       fXi, 1, kC, 0, kBS, tmp, b1 + 8 * kBS, kBS, nullptr, 1.f, 1,
       MT, kBS, kBS, 8, 8);

  // ---- Fourier MLP layer 2 (softshrink):  o2 -> fYr/fYi
  gemm(stream, fXr, kC, 0, kBS, w2t, kBS, 0, kBS * kBS, 0,
       tmp, 0, kC, 0, kBS, nullptr, nullptr, 0, nullptr, 1.f, 0,
       MT, kBS, kBS, 8, 8);
  gemm(stream, fXi, kC, 0, kBS, w2t + 8 * kBS * kBS, kBS, 0, kBS * kBS, 0,
       fYr, 1, kC, 0, kBS, tmp, b2, kBS, nullptr, -1.f, 2,
       MT, kBS, kBS, 8, 8);
  gemm(stream, fXi, kC, 0, kBS, w2t, kBS, 0, kBS * kBS, 0,
       tmp, 0, kC, 0, kBS, nullptr, nullptr, 0, nullptr, 1.f, 0,
       MT, kBS, kBS, 8, 8);
  gemm(stream, fXr, kC, 0, kBS, w2t + 8 * kBS * kBS, kBS, 0, kBS * kBS, 0,
       fYi, 1, kC, 0, kBS, tmp, b2 + 8 * kBS, kBS, nullptr, 1.f, 2,
       MT, kBS, kBS, 8, 8);

  // ---- ifft along H:  Z = G @ o2  per (b,wf)  -> fXr/fXi
  gemm(stream, gr, kH, 0, 0, fYr, sWC, sBfreq, kC, 1,
       tmp, 0, sWC, sBfreq, kC, nullptr, nullptr, 0, nullptr, 1.f, 0,
       kH, kC, kH, kB * kWF, kWF);
  gemm(stream, gi, kH, 0, 0, fYi, sWC, sBfreq, kC, 1,
       fXr, 1, sWC, sBfreq, kC, tmp, nullptr, 0, nullptr, -1.f, 0,
       kH, kC, kH, kB * kWF, kWF);
  gemm(stream, gr, kH, 0, 0, fYi, sWC, sBfreq, kC, 1,
       tmp, 0, sWC, sBfreq, kC, nullptr, nullptr, 0, nullptr, 1.f, 0,
       kH, kC, kH, kB * kWF, kWF);
  gemm(stream, gi, kH, 0, 0, fYr, sWC, sBfreq, kC, 1,
       fXi, 1, sWC, sBfreq, kC, tmp, nullptr, 0, nullptr, 1.f, 0,
       kH, kC, kH, kB * kWF, kWF);

  // ---- irfft along W:  afno = Cr@Zr + Ci@Zi  per (b,h);  M=180 N=768 K=46
  gemm(stream, cr, kWF, 0, 0, fXr, kC, sBfreq, sWC, 1,
       afno, 0, kC, 90 * sRow, sRow, nullptr, nullptr, 0, nullptr, 1.f, 0,
       kW, kC, kWF, kB * kH, kH);
  gemm(stream, ci, kWF, 0, 0, fXi, kC, sBfreq, sWC, 1,
       afno, 0, kC, 90 * sRow, sRow, afno, nullptr, 0, nullptr, 1.f, 0,
       kW, kC, kWF, kB * kH, kH);

  // ---- residual2 = afno + ln1 + x  (-> d_out),  LN2 -> ln2bf
  k_fuse_mid<<<dim3(kT), dim3(256), 0, stream>>>(x, afno, ln1bf, n2w, n2b, out, ln2bf);

  // ---- FC1 + exact GELU:  M=32400 N=3072 K=768
  gemm(stream, ln2bf, kC, 0, 0, fc1t, kC, 0, 0, 0,
       h1, 1, kHID, 0, 0, nullptr, fc1b, 0, nullptr, 1.f, 3,
       kT, kHID, kC, 1, 1);

  // ---- FC2 + bias + residual2 (in place on d_out):  M=32400 N=768 K=3072
  gemm(stream, h1, kHID, 0, 0, fc2t, kHID, 0, 0, 0,
       out, 0, kC, 0, 0, nullptr, fc2b, 0, out, 1.f, 0,
       kT, kC, kHID, 1, 1);
}
